// SimpleStandardAttention_7335804141597
// MI455X (gfx1250) — matrix operand order repaired and checked by reading, not compiled, not run on hardware
//
#include <hip/hip_runtime.h>
#include <hip/hip_bf16.h>

// ---- problem constants (B,T,D,H) = (4,4096,1024,16), HD=64 ----
#define NTOK   16384          // B*T
#define DMODEL 1024
#define NHEAD  16
#define HD     64
#define QKV3   3072           // 3*D

typedef __attribute__((ext_vector_type(16))) __bf16 v16bf;
typedef __attribute__((ext_vector_type(8)))  __bf16 v8bf;
typedef __attribute__((ext_vector_type(8)))  float  v8f;
typedef __attribute__((ext_vector_type(2)))  float  v2f;

typedef __attribute__((ext_vector_type(4))) unsigned int ui32x4;
typedef __attribute__((ext_vector_type(8))) int          i32x8;
typedef __attribute__((ext_vector_type(4))) int          i32x4;

// Scheduling-group control (mask: 0x008=MFMA/WMMA, 0x020=VMEM read, 0x100=DS read)
#if __has_builtin(__builtin_amdgcn_sched_group_barrier)
#define SGB(mask, size, id) __builtin_amdgcn_sched_group_barrier(mask, size, id)
#else
#define SGB(mask, size, id)
#endif

static __device__ __forceinline__ unsigned short f2bf(float f) {
  union { float f; unsigned u; } v; v.f = f;
  unsigned r = v.u + 0x7FFFu + ((v.u >> 16) & 1u);   // round-to-nearest-even
  return (unsigned short)(r >> 16);
}

static __device__ __forceinline__ v8f vzero8() {
  v8f z = {0.f,0.f,0.f,0.f,0.f,0.f,0.f,0.f};
  return z;
}

// ---------------- Kernel 1a: cast x (fp32 -> bf16), 4 elems/thread ----------------
__global__ void __launch_bounds__(256) cast_x_bf16(const float* __restrict__ x,
                                                   unsigned short* __restrict__ xb) {
  int i = (blockIdx.x * 256 + threadIdx.x) * 4;   // n = 16M, exact fit
  float4 v = *(const float4*)(x + i);
  xb[i + 0] = f2bf(v.x);
  xb[i + 1] = f2bf(v.y);
  xb[i + 2] = f2bf(v.z);
  xb[i + 3] = f2bf(v.w);
}

// ---------------- Kernel 1b: transpose + cast W [K][N] fp32 -> Wt [N][K] bf16 ------
__global__ void __launch_bounds__(256) transpose_cast_bf16(const float* __restrict__ src,
                                                           unsigned short* __restrict__ dst,
                                                           int K, int N) {
  int idx = blockIdx.x * 256 + threadIdx.x;       // idx = n*K + k (coalesced stores)
  int n = idx / K;
  int k = idx - n * K;
  dst[idx] = f2bf(src[(size_t)k * N + n]);
}

// ---------------- Kernel 2: fused qkv GEMM + per-token 16x16 head attention -------
// grid.x = NTOK/16 = 1024 ; block = 512 threads = 16 waves ; wave h handles head h.
// dynamic LDS: [16][1024] bf16 x-tile (32KB) + per-head {Q,K,V:16x64 f32, P:16x16 f32}
__global__ void __launch_bounds__(512)
fused_qkv_attn(const unsigned short* __restrict__ xb,      // [NTOK][1024] bf16
               const unsigned short* __restrict__ wqkvT,   // [3072][1024] bf16
               unsigned short* __restrict__ attn_out) {    // [NTOK][1024] bf16
  extern __shared__ char smem[];
  unsigned short* lds_x = (unsigned short*)smem;                 // 32 KB
  float* lds_h = (float*)(smem + 16 * 1024 * 2);                 // 16 * 3328 floats

  const int tile = blockIdx.x;
  const int tid  = threadIdx.x;
  const int wave = tid >> 5;          // head id
  const int lane = tid & 31;
  const int ln16 = lane & 15;
  const bool hi  = (lane >= 16);

  // ---- Phase 1: TDM-load the 16x1024 bf16 x-tile (contiguous 32KB) into LDS ------
  // D# per CDNA5 ISA ch.8: 2D tensor, data_size=2B, one 16384-element row.
  if (wave == 0) {
    unsigned long long ga = (unsigned long long)(xb + (size_t)tile * 16 * DMODEL);
    ui32x4 g0;
    g0.x = 1u;                                             // count=1, user-mode D#
    g0.y = 0u;                                             // lds_addr = 0 (LDS base)
    g0.z = (unsigned)(ga & 0xFFFFFFFFu);                   // global_addr[31:0]
    g0.w = (unsigned)((ga >> 32) & 0x01FFFFFFu) | (2u << 30);  // addr[56:32] | type=2
    i32x8 g1;
    g1[0] = 0x10000;                   // workgroup_mask=0, data_size=1 (2 bytes)
    g1[1] = (int)(16384u << 16);       // tensor_dim0[15:0]=16384 (in [31:16])
    g1[2] = (int)(1u << 16);           // tensor_dim0[31:16]=0 | tensor_dim1[15:0]=1
    g1[3] = (int)(16384u << 16);       // tensor_dim1[31:16]=0 | tile_dim0=16384
    g1[4] = 1;                         // tile_dim1=1 | tile_dim2=0
    g1[5] = 16384;                     // tensor_dim0_stride[31:0]
    g1[6] = 0;                         // stride hi | tensor_dim1_stride lo
    g1[7] = 0;
    i32x4 gz4 = {0, 0, 0, 0};          // groups 2/3 unused (2D tensor)
    i32x8 gz8 = {0, 0, 0, 0, 0, 0, 0, 0};
    __builtin_amdgcn_tensor_load_to_lds(g0, g1, gz4, gz4, gz8, 0);
    __builtin_amdgcn_s_wait_tensorcnt(0);
  }
  __syncthreads();

  // ---- Phase 2: Q,K,V = xtile @ Wqkv[:, head slices]  (bf16 WMMA, K=1024) --------
  // Single base pointer + compile-time immediate offsets per fragment (no pointer
  // array -> no spills). Scheduling groups keep loads one WMMA-batch ahead.
  v8f acc[12];
  #pragma unroll
  for (int i = 0; i < 12; ++i) acc[i] = vzero8();

  // fragment element offset: sec*1048576 + sub*16384 (+k0); byte offsets <= 4.3MB
  const unsigned short* bbase =
      wqkvT + (size_t)(wave * HD + ln16) * 1024 + (hi ? 16 : 0);
  const unsigned short* abase = lds_x + ln16 * 1024 + (hi ? 16 : 0);

  for (int k0 = 0; k0 < DMODEL; k0 += 32) {
    // A fragment (16x32 bf16): lane<16 -> K {k0..k0+15}; lane>=16 -> +16
    v16bf a;
    ((v8bf*)&a)[0] = *(const v8bf*)(abase + k0);
    ((v8bf*)&a)[1] = *(const v8bf*)(abase + k0 + 8);

    v16bf b[12];
    #pragma unroll
    for (int nt = 0; nt < 12; ++nt)
      b[nt] = *(const v16bf*)(bbase + (nt >> 2) * 1048576 + (nt & 3) * 16384 + k0);

    #pragma unroll
    for (int nt = 0; nt < 12; ++nt)
      acc[nt] = __builtin_amdgcn_wmma_f32_16x16x32_bf16(
          false, a, false, b[nt], (short)0, acc[nt], false, false);

    // Enforce: A(ds) | b0-3 | b4-7 | wmma0-3 | b8-11 | wmma4-7 | wmma8-11
    // -> each WMMA batch runs under the next batch's loads (partial loadcnt waits),
    //    peak live B fragments = 8 (64 VGPRs).
    SGB(0x100, 2, 0);   // 2 DS reads (A fragment)
    SGB(0x020, 8, 0);   // b0..b3   (8 global_load_b128)
    SGB(0x020, 8, 0);   // b4..b7
    SGB(0x008, 4, 0);   // wmma b0..b3
    SGB(0x020, 8, 0);   // b8..b11
    SGB(0x008, 4, 0);   // wmma b4..b7
    SGB(0x008, 4, 0);   // wmma b8..b11
  }

  // ---- Phase 3: spill Q,K,V (fp32) to this head's private LDS block --------------
  float* hq = lds_h + wave * 3328;       // 16x64
  float* hk = hq + 16 * HD;              // 16x64
  float* hv = hk + 16 * HD;              // 16x64
  float* hp = hv + 16 * HD;              // 16x16
  #pragma unroll
  for (int nt = 0; nt < 12; ++nt) {
    const int sec = nt >> 2, sub = nt & 3;
    float* base = (sec == 0) ? hq : (sec == 1) ? hk : hv;
    #pragma unroll
    for (int e = 0; e < 8; ++e) {
      const int r = e + (hi ? 8 : 0);                        // C layout row
      base[r * HD + sub * 16 + ln16] = acc[nt][e];
    }
  }

  // ---- Phase 4: S = Q K^T * 1/sqrt(64)  (fp32 WMMA 16x16x4, K=64) ----------------
  v8f s = vzero8();
  #pragma unroll
  for (int k0 = 0; k0 < HD; k0 += 4) {
    v2f aq = *(const v2f*)(hq + ln16 * HD + k0 + (hi ? 2 : 0));   // Q[m][k..k+1]
    v2f bk = *(const v2f*)(hk + ln16 * HD + k0 + (hi ? 2 : 0));   // K^T[k][n]=K[n][k]
    s = __builtin_amdgcn_wmma_f32_16x16x4_f32(
        false, aq, false, bk, (short)0, s, false, false);
  }

  // ---- Phase 5: row softmax over the 16 lanes of each C-layout half --------------
  #pragma unroll
  for (int e = 0; e < 8; ++e) {
    float v = s[e] * 0.125f;                                  // 1/sqrt(HD)
    float m = v;
    #pragma unroll
    for (int msk = 8; msk >= 1; msk >>= 1) m = fmaxf(m, __shfl_xor(m, msk, 32));
    float p = __expf(v - m);
    float sum = p;
    #pragma unroll
    for (int msk = 8; msk >= 1; msk >>= 1) sum += __shfl_xor(sum, msk, 32);
    s[e] = p / sum;
  }

  // ---- Phase 6: O = P @ V  (fp32 WMMA, K=16), store bf16 -------------------------
  #pragma unroll
  for (int e = 0; e < 8; ++e) hp[(e + (hi ? 8 : 0)) * 16 + ln16] = s[e];

  #pragma unroll
  for (int nt = 0; nt < 4; ++nt) {
    v8f oc = vzero8();
    #pragma unroll
    for (int k0 = 0; k0 < 16; k0 += 4) {
      v2f ap = *(const v2f*)(hp + ln16 * 16 + k0 + (hi ? 2 : 0));
      const int kk = k0 + (hi ? 2 : 0);
      v2f bv;
      bv[0] = hv[(kk + 0) * HD + nt * 16 + ln16];
      bv[1] = hv[(kk + 1) * HD + nt * 16 + ln16];
      oc = __builtin_amdgcn_wmma_f32_16x16x4_f32(
          false, ap, false, bv, (short)0, oc, false, false);
    }
    #pragma unroll
    for (int e = 0; e < 8; ++e) {
      const int row = tile * 16 + e + (hi ? 8 : 0);
      const int col = wave * HD + nt * 16 + ln16;
      attn_out[(size_t)row * DMODEL + col] = f2bf(oc[e]);
    }
  }
}

// ---------------- Kernel 3: out = attn_out @ Wout  (bf16 WMMA, fp32 out) ----------
// grid = (NTOK/128, DMODEL/64), block = 256 = 8 waves; wave w: rows [w*16, w*16+16)
__global__ void __launch_bounds__(256)
out_gemm(const unsigned short* __restrict__ ab,      // [NTOK][1024] bf16
         const unsigned short* __restrict__ woutT,   // [1024][1024] bf16 (W^T)
         float* __restrict__ out) {                  // [NTOK][1024] f32
  const int tid  = threadIdx.x;
  const int wave = tid >> 5;
  const int lane = tid & 31;
  const int ln16 = lane & 15;
  const bool hi  = (lane >= 16);
  const int m_base = blockIdx.x * 128 + wave * 16;
  const int n_base = blockIdx.y * 64;

  v8f acc[4];
  #pragma unroll
  for (int i = 0; i < 4; ++i) acc[i] = vzero8();

  const unsigned short* abase = ab + (size_t)(m_base + ln16) * DMODEL + (hi ? 16 : 0);
  const unsigned short* bbase =
      woutT + (size_t)(n_base + ln16) * DMODEL + (hi ? 16 : 0);  // + nt*16384 (imm)

  for (int k0 = 0; k0 < DMODEL; k0 += 32) {
    v16bf a;
    ((v8bf*)&a)[0] = *(const v8bf*)(abase + k0);
    ((v8bf*)&a)[1] = *(const v8bf*)(abase + k0 + 8);

    v16bf b[4];
    #pragma unroll
    for (int nt = 0; nt < 4; ++nt)
      b[nt] = *(const v16bf*)(bbase + nt * 16384 + k0);

    #pragma unroll
    for (int nt = 0; nt < 4; ++nt)
      acc[nt] = __builtin_amdgcn_wmma_f32_16x16x32_bf16(
          false, a, false, b[nt], (short)0, acc[nt], false, false);

    SGB(0x020, 10, 0);  // A (2) + b0..b3 (8) loads first
    SGB(0x008, 4, 0);   // then the 4 WMMAs
  }

  #pragma unroll
  for (int nt = 0; nt < 4; ++nt)
    #pragma unroll
    for (int e = 0; e < 8; ++e) {
      const int row = m_base + e + (hi ? 8 : 0);
      const int col = n_base + nt * 16 + ln16;
      out[(size_t)row * DMODEL + col] = acc[nt][e];
    }
}

// ---------------- launch ----------------------------------------------------------
extern "C" void kernel_launch(void* const* d_in, const int* in_sizes, int n_in,
                              void* d_out, int out_size, void* d_ws, size_t ws_size,
                              hipStream_t stream) {
  const float* x    = (const float*)d_in[0];   // [4,4096,1024]
  const float* Wqkv = (const float*)d_in[1];   // [1024,3072]
  const float* Wout = (const float*)d_in[2];   // [1024,1024]
  float* out = (float*)d_out;                  // [4,4096,1024]

  // workspace layout (bf16 buffers), total ~75.5 MB
  unsigned short* xb    = (unsigned short*)d_ws;                       // 16384*1024
  unsigned short* wqkvT = xb    + (size_t)NTOK * DMODEL;               // 3072*1024
  unsigned short* woutT = wqkvT + (size_t)QKV3 * DMODEL;               // 1024*1024
  unsigned short* attn  = woutT + (size_t)DMODEL * DMODEL;             // 16384*1024

  cast_x_bf16<<<(NTOK * DMODEL) / (256 * 4), 256, 0, stream>>>(x, xb);
  transpose_cast_bf16<<<(QKV3 * DMODEL) / 256, 256, 0, stream>>>(Wqkv, wqkvT, DMODEL, QKV3);
  transpose_cast_bf16<<<(DMODEL * DMODEL) / 256, 256, 0, stream>>>(Wout, woutT, DMODEL, DMODEL);

  const size_t lds_bytes = 16 * 1024 * 2          // x tile (bf16)
                         + 16 * 3328 * 4;         // per-head Q,K,V,P (f32) = 240 KB total
  fused_qkv_attn<<<NTOK / 16, 512, lds_bytes, stream>>>(xb, wqkvT, attn);

  out_gemm<<<dim3(NTOK / 128, DMODEL / 64), 256, 0, stream>>>(attn, woutT, out);
}
